// SparseTF_21552145891656
// MI455X (gfx1250) — matrix operand order repaired and checked by reading, not compiled, not run on hardware
//
#include <hip/hip_runtime.h>
#include <math.h>

#define BATCH     1024
#define INPUT_DIM 30000
#define UNITS     2048

#define BM 128
#define BN 64
#define BK 64

typedef __attribute__((ext_vector_type(16))) __bf16 v16bf;
typedef __attribute__((ext_vector_type(8)))  __bf16 v8bf;
typedef __attribute__((ext_vector_type(4)))  __bf16 v4bf;
typedef __attribute__((ext_vector_type(8)))  float  v8f;
typedef __attribute__((ext_vector_type(4)))  unsigned v4u;
typedef __attribute__((ext_vector_type(8)))  int      v8i;
typedef __attribute__((ext_vector_type(4)))  int      v4i;

#define XB_ELEMS  ((long)BATCH * INPUT_DIM)   /* 30,720,000 bf16 */
#define KBT_ELEMS ((long)UNITS * INPUT_DIM)   /* 61,440,000 bf16 */

// ---------------------------------------------------------------------------
// Kernel 1: x f32 -> Xb bf16 (vectorized).
// ---------------------------------------------------------------------------
__global__ void convert_x_bf16(const float4* __restrict__ x4,
                               __bf16* __restrict__ Xb, long n4) {
    long i = (long)blockIdx.x * blockDim.x + threadIdx.x;
    long stride = (long)gridDim.x * blockDim.x;
    for (; i < n4; i += stride) {
        float4 f = x4[i];
        v4bf b;
        b[0] = (__bf16)f.x; b[1] = (__bf16)f.y;
        b[2] = (__bf16)f.z; b[3] = (__bf16)f.w;
        *(v4bf*)(Xb + 4 * i) = b;
    }
}

// ---------------------------------------------------------------------------
// Kernel 2: zero Kbt (bf16, N-major) with uint4 stores.
// ---------------------------------------------------------------------------
__global__ void zero_u4(uint4* __restrict__ p, long n) {
    long i = (long)blockIdx.x * blockDim.x + threadIdx.x;
    long stride = (long)gridDim.x * blockDim.x;
    uint4 z; z.x = 0u; z.y = 0u; z.z = 0u; z.w = 0u;
    for (; i < n; i += stride) p[i] = z;
}

// ---------------------------------------------------------------------------
// Kernel 3: scatter_nd with duplicate accumulation into bf16 Kbt[c][r],
// via 32-bit CAS (RNE f32->bf16 rounding).
// ---------------------------------------------------------------------------
__global__ void scatter_add_bf16(const long long* __restrict__ ind,
                                 const float* __restrict__ vals,
                                 unsigned* __restrict__ Kw, int nnz) {
    int i = blockIdx.x * blockDim.x + threadIdx.x;
    if (i >= nnz) return;
    long long r = ind[2 * i + 0];
    long long c = ind[2 * i + 1];
    long long idx = c * (long long)INPUT_DIM + r;   // element index in Kbt
    unsigned* word = Kw + (idx >> 1);
    unsigned sh = (idx & 1) ? 16u : 0u;
    float v = vals[i];
    unsigned old = *word, assumed;
    do {
        assumed = old;
        union { unsigned u; float f; } cv;
        cv.u = ((assumed >> sh) & 0xffffu) << 16;    // bf16 -> f32 (exact)
        float nf = cv.f + v;
        unsigned nu; __builtin_memcpy(&nu, &nf, 4);
        nu += 0x7fffu + ((nu >> 16) & 1u);           // round-to-nearest-even
        unsigned nh = (nu >> 16) & 0xffffu;
        unsigned neww = (assumed & ~(0xffffu << sh)) | (nh << sh);
        old = atomicCAS(word, assumed, neww);
    } while (old != assumed);
}

// ---------------------------------------------------------------------------
// TDM: one descriptor moves a rows x tile_k bf16 tile (row stride = strideElems
// elements) from global to LDS. tensor_dim0 = remK gives hardware zero-fill
// beyond the K tail (ISA §8.2: OOB reads return zero).
// D# packing per ISA §8.3 (group 0) / §8.4 (group 1); groups 2-3 zero (2-D).
// 6-arg builtin variant: (v4u g0, v8i g1, v4i g2, v4i g3, v8i, i32 cpol).
// ---------------------------------------------------------------------------
__device__ __forceinline__ void tdm_load_2d(unsigned lds_off, const void* gbase,
                                            unsigned remK, unsigned rows,
                                            unsigned tile_k, unsigned tile_rows,
                                            unsigned strideElems) {
    unsigned long long ga = (unsigned long long)(uintptr_t)gbase;
    v4u g0;
    g0[0] = 1u;                                        // count=1, user D#
    g0[1] = lds_off;                                   // lds_addr (bytes)
    g0[2] = (unsigned)(ga & 0xffffffffu);              // global_addr[31:0]
    g0[3] = (unsigned)((ga >> 32) & 0x01ffffffu)       // global_addr[56:32]
            | (2u << 30);                              // type = 2 ("image")
    v8i g1;
    g1[0] = (int)(1u << 16);                           // data_size=1 -> 2 bytes
    g1[1] = (int)((remK & 0xffffu) << 16);             // tensor_dim0[15:0]
    g1[2] = (int)((remK >> 16) | ((rows & 0xffffu) << 16)); // dim0 hi | dim1 lo
    g1[3] = (int)((rows >> 16) | (tile_k << 16));      // dim1 hi | tile_dim0
    g1[4] = (int)(tile_rows & 0xffffu);                // tile_dim1, tile_dim2=0
    g1[5] = (int)strideElems;                          // tensor_dim0_stride lo32
    g1[6] = 0;                                         // stride hi, dim1_stride
    g1[7] = 0;
    v4i gz4; gz4[0] = 0; gz4[1] = 0; gz4[2] = 0; gz4[3] = 0;
    v8i gz8;
    gz8[0] = 0; gz8[1] = 0; gz8[2] = 0; gz8[3] = 0;
    gz8[4] = 0; gz8[5] = 0; gz8[6] = 0; gz8[7] = 0;
    __builtin_amdgcn_tensor_load_to_lds(g0, g1, gz4, gz4, gz8, 0);
}

// ---------------------------------------------------------------------------
// Kernel 4: out = tanh(Xb @ Kbt^T + bias), bf16 WMMA; tiles staged by the
// Tensor Data Mover, double-buffered, TENSORcnt-synchronized.
// Block 256 threads (8 wave32), tile 128x64, K-step 64.
// ---------------------------------------------------------------------------
__launch_bounds__(256)
__global__ void gemm_tanh_wmma_tdm(const __bf16* __restrict__ Xb,
                                   const __bf16* __restrict__ Kbt,
                                   const float* __restrict__ bias,
                                   float* __restrict__ out) {
    __shared__ __bf16 As[2][BM * BK];   // [m][k], 2 x 16 KB
    __shared__ __bf16 Bs[2][BN * BK];   // [n][k], 2 x  8 KB

    const int tid   = threadIdx.x;
    const int lane  = tid & 31;
    const int wid   = tid >> 5;        // 0..7
    const int m0    = (wid >> 1) * 32; // wave M slab
    const int n0    = (wid & 1) * 32;  // wave N slab
    const int l15   = lane & 15;
    const int hi    = lane >> 4;
    const int mBase = blockIdx.y * BM;
    const int nBase = blockIdx.x * BN;

    const unsigned aB[2] = { (unsigned)(uintptr_t)&As[0][0],
                             (unsigned)(uintptr_t)&As[1][0] };
    const unsigned bB[2] = { (unsigned)(uintptr_t)&Bs[0][0],
                             (unsigned)(uintptr_t)&Bs[1][0] };

    v8f acc[2][2];
#pragma unroll
    for (int mi = 0; mi < 2; ++mi)
#pragma unroll
        for (int ni = 0; ni < 2; ++ni)
#pragma unroll
            for (int e = 0; e < 8; ++e) acc[mi][ni][e] = 0.f;

    const int nk = (INPUT_DIM + BK - 1) / BK;   // 469
    const __bf16* Abase = Xb  + (long)mBase * INPUT_DIM;
    const __bf16* Bbase = Kbt + (long)nBase * INPUT_DIM;

    // prologue: stage tile 0 into buffer 0 (wave 0 only; TDM ignores EXEC,
    // so exactly one wave must issue the tensor ops)
    if (wid == 0) {
        tdm_load_2d(aB[0], Abase, INPUT_DIM, BM, BK, BM, INPUT_DIM);
        tdm_load_2d(bB[0], Bbase, INPUT_DIM, BN, BK, BN, INPUT_DIM);
    }

    for (int kt = 0; kt < nk; ++kt) {
        const int cur = kt & 1;
        if (kt + 1 < nk) {
            const int k1 = (kt + 1) * BK;
            const unsigned remK = (unsigned)(INPUT_DIM - k1);
            if (wid == 0) {
                tdm_load_2d(aB[cur ^ 1], Abase + k1, remK, BM, BK, BM, INPUT_DIM);
                tdm_load_2d(bB[cur ^ 1], Bbase + k1, remK, BN, BK, BN, INPUT_DIM);
            }
            // 2 new TDM ops in flight; <=2 left means current tiles landed
            __builtin_amdgcn_s_wait_tensorcnt(2);
        } else {
            __builtin_amdgcn_s_wait_tensorcnt(0);
        }
        __syncthreads();   // publish wave-0's TDM writes to all waves

        const __bf16* Ab = &As[cur][0];
        const __bf16* Bb = &Bs[cur][0];
#pragma unroll
        for (int ks = 0; ks < BK; ks += 32) {
            v16bf afrag[2];
#pragma unroll
            for (int mi = 0; mi < 2; ++mi) {
                const __bf16* ap = Ab + (m0 + mi * 16 + l15) * BK + ks;
                v8bf lo = *(const v8bf*)(ap + 8 * hi);       // K = 8hi..8hi+7
                v8bf hh = *(const v8bf*)(ap + 16 + 8 * hi);  // K = 16+8hi..+7
#pragma unroll
                for (int e = 0; e < 8; ++e) {
                    afrag[mi][e]     = lo[e];
                    afrag[mi][8 + e] = hh[e];
                }
            }
            v16bf bfrag[2];
#pragma unroll
            for (int ni = 0; ni < 2; ++ni) {
                const __bf16* bp = Bb + (n0 + ni * 16 + l15) * BK + ks;
                v8bf blo = *(const v8bf*)(bp + 8 * hi);       // K = 8hi..8hi+7
                v8bf bhh = *(const v8bf*)(bp + 16 + 8 * hi);  // K = 16+8hi..+7
#pragma unroll
                for (int e = 0; e < 8; ++e) {
                    bfrag[ni][e]     = blo[e];
                    bfrag[ni][8 + e] = bhh[e];
                }
            }
#pragma unroll
            for (int mi = 0; mi < 2; ++mi)
#pragma unroll
                for (int ni = 0; ni < 2; ++ni)
                    acc[mi][ni] = __builtin_amdgcn_wmma_f32_16x16x32_bf16(
                        false, afrag[mi], false, bfrag[ni],
                        (short)0, acc[mi][ni], false, false);
        }
        __syncthreads();   // done reading `cur` before TDM restages it
    }

    // epilogue: C/D layout VGPR r -> row r+8*hi, col l15
#pragma unroll
    for (int mi = 0; mi < 2; ++mi)
#pragma unroll
        for (int ni = 0; ni < 2; ++ni) {
            int n = nBase + n0 + ni * 16 + l15;
            float bv = bias[n];
#pragma unroll
            for (int r = 0; r < 8; ++r) {
                int m = mBase + m0 + mi * 16 + r + 8 * hi;
                out[(long)m * UNITS + n] = tanhf(acc[mi][ni][r] + bv);
            }
        }
}

// ---------------------------------------------------------------------------
// Launch: convert X -> zero Kbt -> bf16 scatter -> TDM-staged WMMA GEMM.
// d_ws layout: [Xb bf16 61.44MB][Kbt bf16 122.88MB] = 184.3MB
// ---------------------------------------------------------------------------
extern "C" void kernel_launch(void* const* d_in, const int* in_sizes, int n_in,
                              void* d_out, int out_size, void* d_ws, size_t ws_size,
                              hipStream_t stream) {
    const float*     x    = (const float*)d_in[0];
    const float*     kv   = (const float*)d_in[1];
    const float*     bias = (const float*)d_in[2];
    const long long* ind  = (const long long*)d_in[3];
    const int        nnz  = in_sizes[1];

    __bf16* Xb  = (__bf16*)d_ws;
    __bf16* Kbt = Xb + XB_ELEMS;
    float*  out = (float*)d_out;

    convert_x_bf16<<<2048, 256, 0, stream>>>((const float4*)x, Xb, XB_ELEMS / 4);

    // zero Kbt: 61,440,000 bf16 = 122,880,000 B = 7,680,000 uint4
    zero_u4<<<4096, 256, 0, stream>>>((uint4*)Kbt, KBT_ELEMS / 8);

    scatter_add_bf16<<<(nnz + 255) / 256, 256, 0, stream>>>(ind, kv,
                                                            (unsigned*)Kbt, nnz);

    dim3 grid(UNITS / BN, BATCH / BM);   // (32, 8)
    gemm_tanh_wmma_tdm<<<grid, 256, 0, stream>>>(Xb, Kbt, bias, out);
}
